// SparseMoE_45526653338242
// MI455X (gfx1250) — matrix operand order repaired and checked by reading, not compiled, not run on hardware
//
#include <hip/hip_runtime.h>
#include <hip/hip_bf16.h>
#include <stdint.h>

// Problem dimensions (fixed by the reference)
#define D_DIM 1024
#define E_DIM 8
#define H_DIM 4096
#define O_DIM 1024
#define N_TOK 4096

// Tiling
#define BM 128
#define BN 128
#define BK 64
#define LDK (BK + 2)          // padded LDS stride (bf16 elems), even -> 4B-aligned pair reads
#define LDA2 (BK + 8)         // gemm2 A stride: 72 elems = 144B row pitch, 16B-aligned + conflict-free
#define MT (N_TOK / BM)       // worst-case m-tiles per expert = 32

#if defined(__gfx1250__) && __has_builtin(__builtin_amdgcn_global_load_async_to_lds_b128)
#define USE_ASYNC_LDS 1
#else
#define USE_ASYNC_LDS 0
#endif

typedef __attribute__((ext_vector_type(16))) __bf16 v16bf;
typedef __attribute__((ext_vector_type(8)))  float  v8f;
typedef __attribute__((ext_vector_type(4)))  int    v4i;

#if USE_ASYNC_LDS
typedef __attribute__((address_space(1))) v4i* gptr_v4i;
typedef __attribute__((address_space(3))) v4i* lptr_v4i;
#endif

union BF16Frag { v16bf v; unsigned short u[16]; uint32_t d[8]; };

__device__ __forceinline__ unsigned short f32_to_bf16(float f) {
  union { float f; uint32_t u; } c; c.f = f;
  uint32_t u = c.u;
  uint32_t r = u + 0x7FFFu + ((u >> 16) & 1u);   // round-to-nearest-even
  if ((u & 0x7FFFFFFFu) > 0x7F800000u) r = u | 0x00400000u; // quiet NaN
  return (unsigned short)(r >> 16);
}

__device__ __forceinline__ void wait_async0() {
#if defined(__gfx1250__)
#if __has_builtin(__builtin_amdgcn_s_wait_asynccnt)
  __builtin_amdgcn_s_wait_asynccnt(0);
#else
  asm volatile("s_wait_asynccnt 0x0" ::: "memory");
#endif
#endif
}

// ---------------------------------------------------------------------------
// Kernel 0: zero output + expert counters
// ---------------------------------------------------------------------------
__global__ void moe_init(float* __restrict__ out, int* __restrict__ counts, int total) {
  int i = blockIdx.x * blockDim.x + threadIdx.x;
  if (i < E_DIM) counts[i] = 0;
  for (int j = i; j < total; j += gridDim.x * blockDim.x) out[j] = 0.0f;
}

// ---------------------------------------------------------------------------
// Kernel 1: router. One wave (32 lanes) per token: logits, softmax, top-2,
// atomic append into per-expert token lists.
// ---------------------------------------------------------------------------
__global__ __launch_bounds__(256) void moe_router(
    const float* __restrict__ x, const float* __restrict__ Wg,
    const float* __restrict__ bg,
    int* __restrict__ counts, int* __restrict__ tok_list, float* __restrict__ tok_w) {
  const int wave = threadIdx.x >> 5;
  const int lane = threadIdx.x & 31;
  const int n = blockIdx.x * 8 + wave;
  if (n >= N_TOK) return;

  float acc[E_DIM];
#pragma unroll
  for (int e = 0; e < E_DIM; ++e) acc[e] = 0.0f;

  const float* xr = x + (size_t)n * D_DIM;
  for (int d = lane; d < D_DIM; d += 32) {
    float xv = xr[d];
    const float* wr = Wg + (size_t)d * E_DIM;
    float4 w0 = *(const float4*)(wr);
    float4 w1 = *(const float4*)(wr + 4);
    acc[0] += xv * w0.x; acc[1] += xv * w0.y; acc[2] += xv * w0.z; acc[3] += xv * w0.w;
    acc[4] += xv * w1.x; acc[5] += xv * w1.y; acc[6] += xv * w1.z; acc[7] += xv * w1.w;
  }
#pragma unroll
  for (int e = 0; e < E_DIM; ++e) {
#pragma unroll
    for (int off = 16; off > 0; off >>= 1)
      acc[e] += __shfl_down(acc[e], off, 32);
  }

  if (lane == 0) {
    float mx = -1e30f;
#pragma unroll
    for (int e = 0; e < E_DIM; ++e) { acc[e] += bg[e]; mx = fmaxf(mx, acc[e]); }
    float sum = 0.0f;
#pragma unroll
    for (int e = 0; e < E_DIM; ++e) { acc[e] = __expf(acc[e] - mx); sum += acc[e]; }
    float inv = 1.0f / sum;

    int b0 = 0;
#pragma unroll
    for (int e = 1; e < E_DIM; ++e) if (acc[e] > acc[b0]) b0 = e;
    int b1 = (b0 == 0) ? 1 : 0;
#pragma unroll
    for (int e = 0; e < E_DIM; ++e) if (e != b0 && acc[e] > acc[b1]) b1 = e;

    int s0 = atomicAdd(&counts[b0], 1);
    tok_list[b0 * N_TOK + s0] = n;
    tok_w[b0 * N_TOK + s0] = acc[b0] * inv;
    int s1 = atomicAdd(&counts[b1], 1);
    tok_list[b1 * N_TOK + s1] = n;
    tok_w[b1 * N_TOK + s1] = acc[b1] * inv;
  }
}

// ---------------------------------------------------------------------------
// Kernel 2: tiny prefix sum over 8 expert counts -> compact h row offsets
// ---------------------------------------------------------------------------
__global__ void moe_offsets(const int* __restrict__ counts, int* __restrict__ offsets) {
  if (threadIdx.x == 0 && blockIdx.x == 0) {
    int s = 0;
    for (int e = 0; e < E_DIM; ++e) { offsets[e] = s; s += counts[e]; }
    offsets[E_DIM] = s;
  }
}

// ---------------------------------------------------------------------------
// Kernel 3: h = relu(x_gathered @ W1[e] + b1[e]) -> bf16 workspace
// ---------------------------------------------------------------------------
__global__ __launch_bounds__(256) void moe_gemm1(
    const float* __restrict__ x, const float* __restrict__ W1,
    const float* __restrict__ b1,
    const int* __restrict__ counts, const int* __restrict__ offsets,
    const int* __restrict__ tok_list, unsigned short* __restrict__ h_ws) {
  __shared__ unsigned short As[BM * LDK];
  __shared__ unsigned short Bt[BN * LDK];
  __shared__ int toks[BM];

  const int e   = blockIdx.x / MT;
  const int mt  = blockIdx.x % MT;
  const int cnt = counts[e];
  const int m0  = mt * BM;
  if (m0 >= cnt) return;
  const int n0   = blockIdx.y * BN;
  const int base = offsets[e];
  const int tid  = threadIdx.x;

  if (tid < BM) {
    int slot = m0 + tid; if (slot > cnt - 1) slot = cnt - 1;
    toks[tid] = tok_list[e * N_TOK + slot];
  }
  __syncthreads();

  const int wave = tid >> 5;
  const int lane = tid & 31;
  const int g    = lane >> 4;
  const int mrow = wave * 16 + (lane & 15);

  v8f acc[8] = {};
  const float* W1e = W1 + (size_t)e * D_DIM * H_DIM;

  for (int k0 = 0; k0 < D_DIM; k0 += BK) {
    { // A tile: gathered token rows, fp32 -> bf16
      int r  = tid >> 1;
      int c0 = (tid & 1) * 32;
      const float* src = x + (size_t)toks[r] * D_DIM + k0 + c0;
      unsigned short* dst = &As[r * LDK + c0];
#pragma unroll
      for (int j = 0; j < 32; j += 4) {
        float4 f = *(const float4*)(src + j);
        *(ushort2*)(dst + j + 0) = make_ushort2(f32_to_bf16(f.x), f32_to_bf16(f.y));
        *(ushort2*)(dst + j + 2) = make_ushort2(f32_to_bf16(f.z), f32_to_bf16(f.w));
      }
    }
    { // B tile: W1 rows k, cols n -> transposed LDS Bt[n][k], fp32 -> bf16
      int k  = tid >> 2;
      int c0 = (tid & 3) * 32;
      const float* src = W1e + (size_t)(k0 + k) * H_DIM + n0 + c0;
#pragma unroll
      for (int j = 0; j < 32; j += 4) {
        float4 f = *(const float4*)(src + j);
        Bt[(c0 + j + 0) * LDK + k] = f32_to_bf16(f.x);
        Bt[(c0 + j + 1) * LDK + k] = f32_to_bf16(f.y);
        Bt[(c0 + j + 2) * LDK + k] = f32_to_bf16(f.z);
        Bt[(c0 + j + 3) * LDK + k] = f32_to_bf16(f.w);
      }
      if (k0 + BK < D_DIM) __builtin_prefetch(src + (size_t)BK * H_DIM, 0, 1);
    }
    __syncthreads();

#pragma unroll
    for (int kk = 0; kk < BK; kk += 32) {
      BF16Frag a;
#pragma unroll
      for (int v = 0; v < 8; ++v) {
        int kl = kk + 2 * v + ((v < 4) ? 0 : 8) + 8 * g;  // ISA 16-bit A layout
        a.d[v] = *(const uint32_t*)&As[mrow * LDK + kl];
      }
#pragma unroll
      for (int nt = 0; nt < 8; ++nt) {
        BF16Frag b;
        int nrow = nt * 16 + (lane & 15);
#pragma unroll
        for (int v = 0; v < 8; ++v) {
          int kl = kk + 2 * v + ((v < 4) ? 0 : 8) + 8 * g; // ISA 16-bit B layout
          b.d[v] = *(const uint32_t*)&Bt[nrow * LDK + kl];
        }
        acc[nt] = __builtin_amdgcn_wmma_f32_16x16x32_bf16(
            false, a.v, false, b.v, (short)0, acc[nt], false, false);
      }
    }
    __syncthreads();
  }

  // epilogue: +bias, relu, store bf16 h
  const float* b1e = b1 + (size_t)e * H_DIM;
#pragma unroll
  for (int nt = 0; nt < 8; ++nt) {
    int col = n0 + nt * 16 + (lane & 15);
    float bv = b1e[col];
#pragma unroll
    for (int r = 0; r < 8; ++r) {
      int row = wave * 16 + r + 8 * g;                    // ISA f32 C/D layout
      if (m0 + row < cnt) {
        float vv = acc[nt][r] + bv;
        vv = vv > 0.0f ? vv : 0.0f;
        h_ws[(size_t)(base + m0 + row) * H_DIM + col] = f32_to_bf16(vv);
      }
    }
  }
}

// ---------------------------------------------------------------------------
// Kernel 4: out += topk_w * (h @ W2[e] + b2[e]), atomic combine over K=2.
// A tile is already bf16 -> use async global->LDS (ASYNCcnt) when available.
// ---------------------------------------------------------------------------
__global__ __launch_bounds__(256) void moe_gemm2(
    const unsigned short* __restrict__ h_ws, const float* __restrict__ W2,
    const float* __restrict__ b2,
    const int* __restrict__ counts, const int* __restrict__ offsets,
    const int* __restrict__ tok_list, const float* __restrict__ tok_w,
    float* __restrict__ out) {
  __shared__ unsigned short As[BM * LDA2];   // 144B row pitch: 16B aligned, conflict-free
  __shared__ unsigned short Bt[BN * LDK];
  __shared__ int   toks[BM];
  __shared__ float tw[BM];

  const int e   = blockIdx.x / MT;
  const int mt  = blockIdx.x % MT;
  const int cnt = counts[e];
  const int m0  = mt * BM;
  if (m0 >= cnt) return;
  const int n0   = blockIdx.y * BN;
  const int base = offsets[e];
  const int tid  = threadIdx.x;

  if (tid < BM) {
    int slot = m0 + tid; if (slot > cnt - 1) slot = cnt - 1;
    toks[tid] = tok_list[e * N_TOK + slot];
    tw[tid]   = tok_w[e * N_TOK + slot];
  }
  __syncthreads();

  const int wave = tid >> 5;
  const int lane = tid & 31;
  const int g    = lane >> 4;
  const int mrow = wave * 16 + (lane & 15);

  v8f acc[8] = {};
  const float* W2e = W2 + (size_t)e * H_DIM * O_DIM;

  for (int k0 = 0; k0 < H_DIM; k0 += BK) {
    { // A tile: bf16 h rows -> LDS
#if USE_ASYNC_LDS
      // 128 rows x 128B = 1024 16-byte chunks; 4 per thread, lanes walk a row.
#pragma unroll
      for (int i = 0; i < 4; ++i) {
        int chunk = tid + 256 * i;
        int r   = chunk >> 3;
        int c16 = chunk & 7;
        int slot = m0 + r; if (slot > cnt - 1) slot = cnt - 1;
        const unsigned short* src =
            h_ws + (size_t)(base + slot) * H_DIM + k0 + c16 * 8;
        __builtin_amdgcn_global_load_async_to_lds_b128(
            (gptr_v4i)src, (lptr_v4i)&As[r * LDA2 + c16 * 8], 0, 0);
      }
#else
      int r  = tid >> 1;
      int c0 = (tid & 1) * 32;
      int slot = m0 + r; if (slot > cnt - 1) slot = cnt - 1;
      const unsigned short* src = h_ws + (size_t)(base + slot) * H_DIM + k0 + c0;
      unsigned short* dst = &As[r * LDA2 + c0];
#pragma unroll
      for (int j = 0; j < 32; j += 8) {
        uint4 q = *(const uint4*)(src + j);
        *(uint32_t*)(dst + j + 0) = q.x;
        *(uint32_t*)(dst + j + 2) = q.y;
        *(uint32_t*)(dst + j + 4) = q.z;
        *(uint32_t*)(dst + j + 6) = q.w;
      }
#endif
    }
    { // B tile: W2 fp32 -> bf16, transposed
      int k  = tid >> 2;
      int c0 = (tid & 3) * 32;
      const float* src = W2e + (size_t)(k0 + k) * O_DIM + n0 + c0;
#pragma unroll
      for (int j = 0; j < 32; j += 4) {
        float4 f = *(const float4*)(src + j);
        Bt[(c0 + j + 0) * LDK + k] = f32_to_bf16(f.x);
        Bt[(c0 + j + 1) * LDK + k] = f32_to_bf16(f.y);
        Bt[(c0 + j + 2) * LDK + k] = f32_to_bf16(f.z);
        Bt[(c0 + j + 3) * LDK + k] = f32_to_bf16(f.w);
      }
      if (k0 + BK < H_DIM) __builtin_prefetch(src + (size_t)BK * O_DIM, 0, 1);
    }
#if USE_ASYNC_LDS
    wait_async0();
#endif
    __syncthreads();

#pragma unroll
    for (int kk = 0; kk < BK; kk += 32) {
      BF16Frag a;
#pragma unroll
      for (int v = 0; v < 8; ++v) {
        int kl = kk + 2 * v + ((v < 4) ? 0 : 8) + 8 * g;
        a.d[v] = *(const uint32_t*)&As[mrow * LDA2 + kl];
      }
#pragma unroll
      for (int nt = 0; nt < 8; ++nt) {
        BF16Frag b;
        int nrow = nt * 16 + (lane & 15);
#pragma unroll
        for (int v = 0; v < 8; ++v) {
          int kl = kk + 2 * v + ((v < 4) ? 0 : 8) + 8 * g;
          b.d[v] = *(const uint32_t*)&Bt[nrow * LDK + kl];
        }
        acc[nt] = __builtin_amdgcn_wmma_f32_16x16x32_bf16(
            false, a.v, false, b.v, (short)0, acc[nt], false, false);
      }
    }
    __syncthreads();
  }

  const float* b2e = b2 + (size_t)e * O_DIM;
#pragma unroll
  for (int nt = 0; nt < 8; ++nt) {
    int col = n0 + nt * 16 + (lane & 15);
    float bv = b2e[col];
#pragma unroll
    for (int r = 0; r < 8; ++r) {
      int row = wave * 16 + r + 8 * g;
      if (m0 + row < cnt) {
        float w = tw[row];
        atomicAdd(&out[(size_t)toks[row] * O_DIM + col], w * (acc[nt][r] + bv));
      }
    }
  }
}

// ---------------------------------------------------------------------------
extern "C" void kernel_launch(void* const* d_in, const int* in_sizes, int n_in,
                              void* d_out, int out_size, void* d_ws, size_t ws_size,
                              hipStream_t stream) {
  (void)in_sizes; (void)n_in; (void)out_size; (void)ws_size;
  const float* x  = (const float*)d_in[0];
  const float* Wg = (const float*)d_in[1];
  const float* bg = (const float*)d_in[2];
  const float* W1 = (const float*)d_in[3];
  const float* b1 = (const float*)d_in[4];
  const float* W2 = (const float*)d_in[5];
  const float* b2 = (const float*)d_in[6];
  float* out = (float*)d_out;

  char* ws = (char*)d_ws;
  int*   counts   = (int*)(ws + 0);
  int*   offsets  = (int*)(ws + 64);
  int*   tok_list = (int*)(ws + 256);
  size_t off_tw   = 256 + (size_t)E_DIM * N_TOK * 4;
  float* tok_w    = (float*)(ws + off_tw);
  size_t off_h    = (off_tw + (size_t)E_DIM * N_TOK * 4 + 255) & ~(size_t)255;
  unsigned short* h_ws = (unsigned short*)(ws + off_h);   // 2*N_TOK rows x H_DIM bf16 = 64 MB

  hipLaunchKernelGGL(moe_init, dim3(2048), dim3(256), 0, stream,
                     out, counts, N_TOK * O_DIM);
  hipLaunchKernelGGL(moe_router, dim3(N_TOK / 8), dim3(256), 0, stream,
                     x, Wg, bg, counts, tok_list, tok_w);
  hipLaunchKernelGGL(moe_offsets, dim3(1), dim3(1), 0, stream, counts, offsets);
  hipLaunchKernelGGL(moe_gemm1, dim3(E_DIM * MT, H_DIM / BN), dim3(256), 0, stream,
                     x, W1, b1, counts, offsets, tok_list, h_ws);
  hipLaunchKernelGGL(moe_gemm2, dim3(E_DIM * MT, O_DIM / BN), dim3(256), 0, stream,
                     h_ws, W2, b2, counts, offsets, tok_list, tok_w, out);
}
